// GraphGNN_24275155157311
// MI455X (gfx1250) — matrix operand order repaired and checked by reading, not compiled, not run on hardware
//
#include <hip/hip_runtime.h>
#include <hip/hip_bf16.h>
#include <hip/hip_fp16.h>

typedef __attribute__((ext_vector_type(16))) _Float16 v16h;
typedef __attribute__((ext_vector_type(8)))  float    v8f;

#define B_    8
#define N_    10000
#define E_    320000
#define INDIM 16
#define EH    32
#define EOUT  32
#define ODIM  32
#define DEDGE 35

#define AGG_FLOATS   (B_*N_*ODIM)               /* 2,560,000 */
#define STATS_ACC_OFF (AGG_FLOATS)              /* 4 floats: s0,s1,q0,q1 */
#define STATS_FIN_OFF (AGG_FLOATS + 4)          /* 4 floats: m0,m1,rstd0,rstd1 */
#define FRAGS_BYTE_OFF ((size_t)(AGG_FLOATS + 8) * 4)  /* 10,240,032 B, 32B aligned */

// fast sigmoid: v_exp_f32 + v_rcp_f32, no div_scale/div_fixup sequence
__device__ __forceinline__ float sigmoidf_(float v) {
    return __builtin_amdgcn_rcpf(1.0f + __expf(-v));
}

__device__ __forceinline__ v8f wmma_f16(v16h a, v16h b, v8f c) {
    return __builtin_amdgcn_wmma_f32_16x16x32_f16(false, a, false, b,
                                                  (short)0, c, false, false);
}

// ---------------------------------------------------------------- zero
__global__ void k_zero(float* __restrict__ p, int n) {
    int i = blockIdx.x * blockDim.x + threadIdx.x;
    int stride = gridDim.x * blockDim.x;
    for (; i < n; i += stride) p[i] = 0.0f;
}

// ---------------------------------------------------------------- edge_attr stats
__global__ __launch_bounds__(256) void k_stats(const float* __restrict__ ea,
                                               float* __restrict__ acc) {
    __shared__ float sm[4][256];
    float s0 = 0.f, s1 = 0.f, q0 = 0.f, q1 = 0.f;
    for (int e = blockIdx.x * blockDim.x + threadIdx.x; e < E_;
         e += gridDim.x * blockDim.x) {
        float a0 = ea[2 * e + 0], a1 = ea[2 * e + 1];
        s0 += a0; s1 += a1; q0 += a0 * a0; q1 += a1 * a1;
    }
    int t = threadIdx.x;
    sm[0][t] = s0; sm[1][t] = s1; sm[2][t] = q0; sm[3][t] = q1;
    __syncthreads();
    for (int off = 128; off > 0; off >>= 1) {
        if (t < off) {
            #pragma unroll
            for (int j = 0; j < 4; j++) sm[j][t] += sm[j][t + off];
        }
        __syncthreads();
    }
    if (t < 4) atomicAdd(&acc[t], sm[t][0]);
}

// ---------------------------------------------------------------- weight fragment packing
// B-matrix 32x16 (KxN) f16 layout: lanes 0-15 -> N=lane, K=0..7 / 16..23 in 16 halves;
// lanes 16-31 -> same N, K=8..15 / 24..31.  Fragment f: [f*32 + lane] * 16 halves.
// frags 0..3: W1 (kc = f>>1, nc = f&1, Kactual=35, padded to 64)
// frags 4..5: W2 (nc = f-4), frags 6..7: W3 (nc = f-6)
__global__ __launch_bounds__(256) void k_prep(const float* __restrict__ W1,
                                              const float* __restrict__ W2,
                                              const float* __restrict__ W3,
                                              const float* __restrict__ acc,
                                              float* __restrict__ fin,
                                              _Float16* __restrict__ frags) {
    int t = threadIdx.x;
    if (t == 0) {
        float inv = 1.0f / (float)E_;
        float m0 = acc[0] * inv, m1 = acc[1] * inv;
        float v0 = (acc[2] - acc[0] * acc[0] * inv) / (float)(E_ - 1);
        float v1 = (acc[3] - acc[1] * acc[1] * inv) / (float)(E_ - 1);
        fin[0] = m0; fin[1] = m1; fin[2] = rsqrtf(v0); fin[3] = rsqrtf(v1);
    }
    int frag = t >> 5, lane = t & 31;
    int nl = lane & 15, hi = lane >> 4;
    const float* W; int Kact, kc = 0, nc;
    if (frag < 4)      { W = W1; Kact = DEDGE; kc = frag >> 1; nc = frag & 1; }
    else if (frag < 6) { W = W2; Kact = EH;    nc = frag - 4; }
    else               { W = W3; Kact = EOUT;  nc = frag - 6; }
    int n = nc * 16 + nl;
    _Float16* dst = frags + ((size_t)(frag * 32 + lane)) * 16;
    #pragma unroll
    for (int j = 0; j < 16; j++) {
        int k = kc * 32 + ((j < 8) ? (hi * 8 + j) : (16 + hi * 8 + (j - 8)));
        float v = (k < Kact) ? W[n * Kact + k] : 0.0f;
        dst[j] = (_Float16)v;
    }
}

// ---------------------------------------------------------------- edge MLP + scatter
__global__ __launch_bounds__(256) void k_edge(
    const float* __restrict__ x, const float* __restrict__ ea,
    const float* __restrict__ wmean, const float* __restrict__ wstd,
    const float* __restrict__ b1, const float* __restrict__ b2,
    const int* __restrict__ eidx, const float* __restrict__ stats,
    const _Float16* __restrict__ frags, float* __restrict__ agg) {

    __shared__ _Float16 sm[8][16][32];   // per-wave 16edges x 32hidden
    const int wave = threadIdx.x >> 5;
    const int lane = threadIdx.x & 31;
    const int gid  = blockIdx.x * 8 + wave;
    const int b    = gid / (E_ / 16);
    const int et   = (gid % (E_ / 16)) * 16;
    const int nl = lane & 15, hi = lane >> 4;

    const int e   = et + nl;
    const int srcn = eidx[e];
    const int tgtn = eidx[E_ + e];
    const float* xs = x + ((size_t)b * N_ + srcn) * INDIM;
    const float* xt = x + ((size_t)b * N_ + tgtn) * INDIM;

    // A-fragment layout (16-bit A 16x32): lane<16 -> M=lane, K 0..7 / 16..23
    //                                     lane>=16 -> M=lane-16, K 8..15 / 24..31
    v16h a0, a1 = {};
    if (hi == 0) {
        const float4 s0 = *(const float4*)(xs);
        const float4 s1 = *(const float4*)(xs + 4);
        const float4 t0 = *(const float4*)(xt);
        const float4 t1 = *(const float4*)(xt + 4);
        a0[0]=(_Float16)s0.x; a0[1]=(_Float16)s0.y; a0[2]=(_Float16)s0.z; a0[3]=(_Float16)s0.w;
        a0[4]=(_Float16)s1.x; a0[5]=(_Float16)s1.y; a0[6]=(_Float16)s1.z; a0[7]=(_Float16)s1.w;
        a0[8]=(_Float16)t0.x; a0[9]=(_Float16)t0.y; a0[10]=(_Float16)t0.z; a0[11]=(_Float16)t0.w;
        a0[12]=(_Float16)t1.x; a0[13]=(_Float16)t1.y; a0[14]=(_Float16)t1.z; a0[15]=(_Float16)t1.w;
        // edge weight from src wind (x[...,14:16]) + edge_attr
        const float2 wv  = *(const float2*)(xs + 14);
        const float2 eav = *(const float2*)(ea + 2 * e);
        float speed = wv.x * wstd[0] + wmean[0];
        float direc = wv.y * wstd[1] + wmean[1];
        float theta = fabsf(eav.y - direc);
        float w = fmaxf(0.0f,
                        3.0f * speed * __cosf(theta * 22.5f) *
                        __builtin_amdgcn_rcpf(eav.x));
        a1[0] = (_Float16)((eav.x - stats[0]) * stats[2]);
        a1[1] = (_Float16)((eav.y - stats[1]) * stats[3]);
        a1[2] = (_Float16)w;
    } else {
        const float4 s0 = *(const float4*)(xs + 8);
        const float4 s1 = *(const float4*)(xs + 12);
        const float4 t0 = *(const float4*)(xt + 8);
        const float4 t1 = *(const float4*)(xt + 12);
        a0[0]=(_Float16)s0.x; a0[1]=(_Float16)s0.y; a0[2]=(_Float16)s0.z; a0[3]=(_Float16)s0.w;
        a0[4]=(_Float16)s1.x; a0[5]=(_Float16)s1.y; a0[6]=(_Float16)s1.z; a0[7]=(_Float16)s1.w;
        a0[8]=(_Float16)t0.x; a0[9]=(_Float16)t0.y; a0[10]=(_Float16)t0.z; a0[11]=(_Float16)t0.w;
        a0[12]=(_Float16)t1.x; a0[13]=(_Float16)t1.y; a0[14]=(_Float16)t1.z; a0[15]=(_Float16)t1.w;
        // K 40..47 / 56..63 are padding -> zeros
    }

    const v16h* F = (const v16h*)frags;
    const v16h B00 = F[0 * 32 + lane];
    const v16h B01 = F[1 * 32 + lane];
    const v16h B10 = F[2 * 32 + lane];
    const v16h B11 = F[3 * 32 + lane];

    v8f c0 = {}, c1 = {};
    c0 = wmma_f16(a0, B00, c0);
    c0 = wmma_f16(a1, B10, c0);
    c1 = wmma_f16(a0, B01, c1);
    c1 = wmma_f16(a1, B11, c1);

    // sigmoid + bias, spill 16x32 tile to LDS (C layout -> row-major)
    const float bb0 = b1[nl], bb1 = b1[nl + 16];
    #pragma unroll
    for (int r = 0; r < 8; r++) {
        sm[wave][hi * 8 + r][nl]      = (_Float16)sigmoidf_(c0[r] + bb0);
        sm[wave][hi * 8 + r][nl + 16] = (_Float16)sigmoidf_(c1[r] + bb1);
    }
    __syncthreads();

    // reload in A layout: M=nl, K per lane half
    v16h a2;
    #pragma unroll
    for (int j = 0; j < 16; j++) {
        int k = (j < 8) ? (hi * 8 + j) : (16 + hi * 8 + (j - 8));
        a2[j] = sm[wave][nl][k];
    }

    const v16h B20 = F[4 * 32 + lane];
    const v16h B21 = F[5 * 32 + lane];
    v8f d0 = {}, d1 = {};
    d0 = wmma_f16(a2, B20, d0);
    d1 = wmma_f16(a2, B21, d1);

    // sigmoid + bias, scatter-add into agg[b, tgt, :]
    const float cb0 = b2[nl], cb1 = b2[nl + 16];
    const int* tp = eidx + E_ + et + hi * 8;
    const int4 tA = *(const int4*)tp;
    const int4 tB = *(const int4*)(tp + 4);
    const int tg[8] = {tA.x, tA.y, tA.z, tA.w, tB.x, tB.y, tB.z, tB.w};
    #pragma unroll
    for (int r = 0; r < 8; r++) {
        float* basep = agg + ((size_t)b * N_ + tg[r]) * ODIM;
        atomicAdd(basep + nl,      sigmoidf_(d0[r] + cb0));
        atomicAdd(basep + nl + 16, sigmoidf_(d1[r] + cb1));
    }
}

// ---------------------------------------------------------------- node layer
__global__ __launch_bounds__(256) void k_node(const float* __restrict__ agg,
                                              const float* __restrict__ b3,
                                              const _Float16* __restrict__ frags,
                                              float* __restrict__ out) {
    const int wave = threadIdx.x >> 5, lane = threadIdx.x & 31;
    const int gid = blockIdx.x * 8 + wave;
    const int b  = gid / (N_ / 16);
    const int nt = (gid % (N_ / 16)) * 16;
    const int nl = lane & 15, hi = lane >> 4;

    const float* ap = agg + ((size_t)b * N_ + nt + nl) * ODIM;
    const float4 g0 = *(const float4*)(ap + hi * 8);
    const float4 g1 = *(const float4*)(ap + hi * 8 + 4);
    const float4 g2 = *(const float4*)(ap + 16 + hi * 8);
    const float4 g3 = *(const float4*)(ap + 16 + hi * 8 + 4);
    v16h a;
    a[0]=(_Float16)g0.x; a[1]=(_Float16)g0.y; a[2]=(_Float16)g0.z; a[3]=(_Float16)g0.w;
    a[4]=(_Float16)g1.x; a[5]=(_Float16)g1.y; a[6]=(_Float16)g1.z; a[7]=(_Float16)g1.w;
    a[8]=(_Float16)g2.x; a[9]=(_Float16)g2.y; a[10]=(_Float16)g2.z; a[11]=(_Float16)g2.w;
    a[12]=(_Float16)g3.x; a[13]=(_Float16)g3.y; a[14]=(_Float16)g3.z; a[15]=(_Float16)g3.w;

    const v16h* F = (const v16h*)frags;
    v8f c0 = {}, c1 = {};
    c0 = wmma_f16(a, F[6 * 32 + lane], c0);
    c1 = wmma_f16(a, F[7 * 32 + lane], c1);

    const float bb0 = b3[nl], bb1 = b3[nl + 16];
    float* op = out + ((size_t)b * N_ + nt + hi * 8) * ODIM;
    #pragma unroll
    for (int r = 0; r < 8; r++) {
        op[r * ODIM + nl]      = sigmoidf_(c0[r] + bb0);
        op[r * ODIM + nl + 16] = sigmoidf_(c1[r] + bb1);
    }
}

// ---------------------------------------------------------------- launch
extern "C" void kernel_launch(void* const* d_in, const int* in_sizes, int n_in,
                              void* d_out, int out_size, void* d_ws, size_t ws_size,
                              hipStream_t stream) {
    const float* x     = (const float*)d_in[0];
    const float* ea    = (const float*)d_in[1];
    const float* wmean = (const float*)d_in[2];
    const float* wstd  = (const float*)d_in[3];
    const float* W1    = (const float*)d_in[4];
    const float* b1    = (const float*)d_in[5];
    const float* W2    = (const float*)d_in[6];
    const float* b2    = (const float*)d_in[7];
    const float* W3    = (const float*)d_in[8];
    const float* b3    = (const float*)d_in[9];
    const int*   eidx  = (const int*)d_in[10];
    float* out = (float*)d_out;

    float* ws   = (float*)d_ws;
    float* agg  = ws;
    float* acc  = ws + STATS_ACC_OFF;
    float* fin  = ws + STATS_FIN_OFF;
    _Float16* frags = (_Float16*)((char*)d_ws + FRAGS_BYTE_OFF);

    k_zero <<<2048, 256, 0, stream>>>(agg, AGG_FLOATS + 4);
    k_stats<<<256,  256, 0, stream>>>(ea, acc);
    k_prep <<<1,    256, 0, stream>>>(W1, W2, W3, acc, fin, frags);
    k_edge <<<(B_ * (E_ / 16)) / 8, 256, 0, stream>>>(x, ea, wmean, wstd, b1, b2,
                                                      eidx, fin, frags, agg);
    k_node <<<(B_ * (N_ / 16)) / 8, 256, 0, stream>>>(agg, b3, frags, out);
}
